// LlamaAttentionQKV_37409165148605
// MI455X (gfx1250) — matrix operand order repaired and checked by reading, not compiled, not run on hardware
//
#include <hip/hip_runtime.h>
#include <hip/hip_bf16.h>
#include <math.h>

typedef __attribute__((ext_vector_type(16))) _Float16 v16h;
typedef __attribute__((ext_vector_type(8)))  float    v8f;

#define B_    2
#define S_    1024
#define H_    32
#define HKV_  8
#define D_    128
#define G_    4      // GQA group = H/HKV
#define QBLK  32     // q rows per workgroup
#define KBLK  32     // kv rows per iteration

#define QSTR  136    // halves; row = 272B = 17*16 -> 16B aligned
#define KSTR  136
#define VTS   40     // halves; row = 80B -> 16B aligned
#define PSTR  40

// shared-memory overlay (bytes):
//   [0 .. 37888)  : prologue: RoPE'd Q (34816B); then K dbl buf (2*8704) + V^T dbl buf (2*10240)
//   [37888..48128): per-wave P relayout (8*16*40*2)
#define KBUF_OFF  0
#define KBUF_SZ   8704
#define VBUF_OFF  17408
#define VBUF_SZ   10240
#define PBUF_OFF  37888
#define SHM_BYTES 48128

#define LOG2E 1.44269504088896340736f
#define MASK2 (-1.442695e9f)   // -1e9 premultiplied by log2(e) (base-2 softmax domain)

// ds_swizzle xor-butterfly (group-of-32 mode: offset = xor<<10 | 0x1f)
#define SWZ_MAXF(t, imm) \
  t = fmaxf(t, __int_as_float(__builtin_amdgcn_ds_swizzle(__float_as_int(t), imm)))

__device__ __forceinline__ v16h ld_frag_8x2(const _Float16* p0, const _Float16* p1) {
  v16h r;
#pragma unroll
  for (int i = 0; i < 8; ++i) { r[i] = p0[i]; r[8 + i] = p1[i]; }
  return r;
}

__device__ __forceinline__ v16h ld_frag_16(const _Float16* p) {
  v16h r;
#pragma unroll
  for (int i = 0; i < 16; ++i) r[i] = p[i];
  return r;
}

// RoPE (Llama rotate_half) one row: d0 in [part*32, part*32+32) plus +64 partners.
__device__ __forceinline__ void stage_rope_row(const float* __restrict__ xrow,
                                               const float* __restrict__ crow,
                                               const float* __restrict__ srow,
                                               _Float16* __restrict__ dst,
                                               int part) {
#pragma unroll
  for (int dd = 0; dd < 32; dd += 4) {
    const int d0 = part * 32 + dd;
    const float4 x0 = *(const float4*)(xrow + d0);
    const float4 x1 = *(const float4*)(xrow + d0 + 64);
    const float4 c0 = *(const float4*)(crow + d0);
    const float4 s0 = *(const float4*)(srow + d0);
    const float4 c1 = *(const float4*)(crow + d0 + 64);
    const float4 s1 = *(const float4*)(srow + d0 + 64);
    dst[d0 + 0] = (_Float16)(x0.x * c0.x - x1.x * s0.x);
    dst[d0 + 1] = (_Float16)(x0.y * c0.y - x1.y * s0.y);
    dst[d0 + 2] = (_Float16)(x0.z * c0.z - x1.z * s0.z);
    dst[d0 + 3] = (_Float16)(x0.w * c0.w - x1.w * s0.w);
    dst[d0 + 64 + 0] = (_Float16)(x1.x * c1.x + x0.x * s1.x);
    dst[d0 + 64 + 1] = (_Float16)(x1.y * c1.y + x0.y * s1.y);
    dst[d0 + 64 + 2] = (_Float16)(x1.z * c1.z + x0.z * s1.z);
    dst[d0 + 64 + 3] = (_Float16)(x1.w * c1.w + x0.w * s1.w);
  }
}

// One 32-wide KV block. O[0..7] = output tiles over D; O[8] = P*onescol tile
// whose column 0 accumulates the softmax denominator (ones-column trick, no
// cross-lane row-sum reductions needed).
template <bool MASKED>
__device__ __forceinline__ void attn_block(const v16h* __restrict__ qf,
                                           v8f* __restrict__ O,            // 9 tiles
                                           float* __restrict__ m_run,
                                           const _Float16* __restrict__ kbuf,
                                           const _Float16* __restrict__ vbuf,
                                           _Float16* __restrict__ pbuf,
                                           v16h onesf,
                                           int jbase, int qrow0, int n, int sel,
                                           float scale2) {
  v8f S0, S1;
#pragma unroll
  for (int r = 0; r < 8; ++r) { S0[r] = 0.0f; S1[r] = 0.0f; }
#pragma unroll
  for (int kk = 0; kk < 4; ++kk) {
    const v16h b0 = ld_frag_16(kbuf + n * KSTR + kk * 32 + sel * 16);
    const v16h b1 = ld_frag_16(kbuf + (16 + n) * KSTR + kk * 32 + sel * 16);
    S0 = __builtin_amdgcn_wmma_f32_16x16x32_f16(false, qf[kk], false, b0,
                                                (short)0, S0, false, false);
    S1 = __builtin_amdgcn_wmma_f32_16x16x32_f16(false, qf[kk], false, b1,
                                                (short)0, S1, false, false);
  }

  float al[8];
#pragma unroll
  for (int r = 0; r < 8; ++r) {
    float s0 = S0[r] * scale2;     // base-2 softmax domain
    float s1 = S1[r] * scale2;
    if (MASKED) {
      const int qpos = qrow0 + r;
      s0 += ((jbase + n)      <= qpos) ? 0.0f : MASK2;
      s1 += ((jbase + 16 + n) <= qpos) ? 0.0f : MASK2;
    }
    float t = fmaxf(s0, s1);
    SWZ_MAXF(t, 0x041f);  // xor 1
    SWZ_MAXF(t, 0x081f);  // xor 2
    SWZ_MAXF(t, 0x101f);  // xor 4
    SWZ_MAXF(t, 0x201f);  // xor 8
    const float mnew = fmaxf(m_run[r], t);
    al[r] = __builtin_amdgcn_exp2f(m_run[r] - mnew);
    m_run[r] = mnew;
    const float p0 = __builtin_amdgcn_exp2f(s0 - mnew);
    const float p1 = __builtin_amdgcn_exp2f(s1 - mnew);
    pbuf[(sel * 8 + r) * PSTR + n]      = (_Float16)p0;
    pbuf[(sel * 8 + r) * PSTR + 16 + n] = (_Float16)p1;
  }
#pragma unroll
  for (int c = 0; c < 9; ++c)
#pragma unroll
    for (int r = 0; r < 8; ++r) O[c][r] *= al[r];

  const _Float16* pr = pbuf + n * PSTR;
  const v16h pf = ld_frag_16(pr + sel * 16);
#pragma unroll
  for (int c = 0; c < 8; ++c) {
    const v16h vf = ld_frag_16(vbuf + (c * 16 + n) * VTS + sel * 16);
    O[c] = __builtin_amdgcn_wmma_f32_16x16x32_f16(false, pf, false, vf,
                                                  (short)0, O[c], false, false);
  }
  O[8] = __builtin_amdgcn_wmma_f32_16x16x32_f16(false, pf, false, onesf,
                                                (short)0, O[8], false, false);
}

__global__ __launch_bounds__(256)
void llama_attn_fwd_kernel(const float* __restrict__ q,
                           const float* __restrict__ k,
                           const float* __restrict__ v,
                           const float* __restrict__ cosT,
                           const float* __restrict__ sinT,
                           const float* __restrict__ scale_p,
                           float* __restrict__ out) {
  __shared__ alignas(16) unsigned char shm[SHM_BYTES];

  const int tid  = threadIdx.x;
  const int lane = tid & 31;
  const int wid  = tid >> 5;

  const int q0  = blockIdx.x * QBLK;
  const int hkv = blockIdx.y;
  const int b   = blockIdx.z;

  const int g     = wid >> 1;
  const int qsub  = wid & 1;
  const int h     = hkv * G_ + g;
  const int qbase = q0 + qsub * 16;

  const int n   = lane & 15;
  const int sel = lane >> 4;

  const float scale2 = scale_p[0] * LOG2E;

  // B fragment of the ones-column (col 0 = 1, rest 0): per-lane splat
  v16h onesf;
  {
    const _Float16 ov = (n == 0) ? (_Float16)1.0f : (_Float16)0.0f;
#pragma unroll
    for (int i = 0; i < 16; ++i) onesf[i] = ov;
  }

  // ---- stage Q (RoPE -> f16 -> LDS), 256 tasks ----
  {
    const int part = tid & 1;
    const int row  = (tid >> 1) & 31;
    const int gh   = tid >> 6;
    const int pos  = q0 + row;
    const float* qrow = q + ((size_t)(b * S_ + pos) * H_ + (hkv * G_ + gh)) * D_;
    stage_rope_row(qrow, cosT + (size_t)pos * D_, sinT + (size_t)pos * D_,
                   (_Float16*)shm + (gh * QBLK + row) * QSTR, part);
  }
  __syncthreads();

  // Q A-fragments -> registers for the whole kernel
  v16h qf[4];
  {
    const _Float16* qr = (const _Float16*)shm + (g * QBLK + qsub * 16 + n) * QSTR;
#pragma unroll
    for (int kk = 0; kk < 4; ++kk) {
      const _Float16* p0 = qr + kk * 32 + sel * 16;
      qf[kk] = ld_frag_16(p0);
    }
  }
  __syncthreads();  // Q region free: reuse for K/V double buffers

  auto stage_kv = [&](int jbase, int buf) {
    _Float16* kd = (_Float16*)(shm + KBUF_OFF + buf * KBUF_SZ);
    _Float16* vd = (_Float16*)(shm + VBUF_OFF + buf * VBUF_SZ);
    if (tid < 64) {
      const int part = tid & 1;
      const int row  = tid >> 1;
      const int pos  = jbase + row;
      const float* krow = k + ((size_t)(b * S_ + pos) * HKV_ + hkv) * D_;
      stage_rope_row(krow, cosT + (size_t)pos * D_, sinT + (size_t)pos * D_,
                     kd + row * KSTR, part);
    }
    {
      const int p8  = tid & 7;
      const int row = tid >> 3;
      const float* vrow = v + ((size_t)(b * S_ + jbase + row) * HKV_ + hkv) * D_;
#pragma unroll
      for (int dd = 0; dd < 16; dd += 4) {
        const int d = p8 * 16 + dd;
        const float4 xv = *(const float4*)(vrow + d);
        vd[(d + 0) * VTS + row] = (_Float16)xv.x;
        vd[(d + 1) * VTS + row] = (_Float16)xv.y;
        vd[(d + 2) * VTS + row] = (_Float16)xv.z;
        vd[(d + 3) * VTS + row] = (_Float16)xv.w;
      }
    }
  };

  // running max (base-2 domain) + 9 accumulator tiles (8 output + 1 denominator)
  float m_run[8];
  v8f   O[9];
#pragma unroll
  for (int r = 0; r < 8; ++r) m_run[r] = -3.0e38f;
#pragma unroll
  for (int c = 0; c < 9; ++c)
#pragma unroll
    for (int r = 0; r < 8; ++r) O[c][r] = 0.0f;

  const int jmax  = q0 >> 5;
  const int qrow0 = qbase + sel * 8;
  _Float16* const pbuf = (_Float16*)(shm + PBUF_OFF) + wid * 16 * PSTR;

  stage_kv(0, 0);
  __syncthreads();

  for (int j = 0; j <= jmax; ++j) {
    const int cur = j & 1;
    if (j < jmax) stage_kv((j + 1) * KBLK, cur ^ 1);  // overlap loads with compute
    if (j + 2 <= jmax) {                              // keep L2 warm 2 blocks ahead
      const size_t nb = ((size_t)(b * S_ + (j + 2) * KBLK) * HKV_ + hkv) * D_;
      __builtin_prefetch(k + nb + (tid & 127) * 2, 0, 1);
      __builtin_prefetch(v + nb + (tid & 127) * 2, 0, 1);
    }

    const _Float16* kbuf = (const _Float16*)(shm + KBUF_OFF + cur * KBUF_SZ);
    const _Float16* vbuf = (const _Float16*)(shm + VBUF_OFF + cur * VBUF_SZ);
    if (j < jmax) {
      attn_block<false>(qf, O, m_run, kbuf, vbuf, pbuf, onesf,
                        j * KBLK, qrow0, n, sel, scale2);
    } else {
      attn_block<true>(qf, O, m_run, kbuf, vbuf, pbuf, onesf,
                       j * KBLK, qrow0, n, sel, scale2);
    }
    __syncthreads();
  }

  // ---- normalize by denominator (O[8] col 0, broadcast within each half) ----
  float linv[8];
#pragma unroll
  for (int r = 0; r < 8; ++r) {
    // BCASTX16: src_lane = lane & 0x10  (lane 0 -> low half, lane 16 -> high half)
    const float lsum = __int_as_float(
        __builtin_amdgcn_ds_swizzle(__float_as_int(O[8][r]), 0x0010));
    linv[r] = 1.0f / lsum;
  }
#pragma unroll
  for (int c = 0; c < 8; ++c) {
#pragma unroll
    for (int r = 0; r < 8; ++r) {
      const int qq = qbase + sel * 8 + r;
      out[((size_t)(b * S_ + qq) * H_ + h) * D_ + c * 16 + n] = O[c][r] * linv[r];
    }
  }
}

extern "C" void kernel_launch(void* const* d_in, const int* in_sizes, int n_in,
                              void* d_out, int out_size, void* d_ws, size_t ws_size,
                              hipStream_t stream) {
  (void)in_sizes; (void)n_in; (void)d_ws; (void)ws_size; (void)out_size;
  const float* q     = (const float*)d_in[0];
  const float* k     = (const float*)d_in[1];
  const float* v     = (const float*)d_in[2];
  const float* cosT  = (const float*)d_in[3];
  const float* sinT  = (const float*)d_in[4];
  // d_in[5] = attention_mask: pure causal, computed inline in the kernel
  const float* scale = (const float*)d_in[6];
  float* out = (float*)d_out;

  dim3 grid(S_ / QBLK, HKV_, B_);
  llama_attn_fwd_kernel<<<grid, 256, 0, stream>>>(q, k, v, cosT, sinT, scale, out);
}
